// GCN_GAT_86199993631182
// MI455X (gfx1250) — matrix operand order repaired and checked by reading, not compiled, not run on hardware
//
#include <hip/hip_runtime.h>
#include <hip/hip_bf16.h>
#include <math.h>

// ---------------- problem constants ----------------
constexpr int Bc = 4, Tc = 32, Nc = 370, Ec = 11840;
constexpr int Hc = 64, HEADSc = 4, DHc = 16, Lc = 3, Kc = 5;
constexpr int Gc  = Bc * Tc;            // 128 graphs
constexpr int E2c = Ec + Nc;            // 12210 edges incl self loops
constexpr int GNc = Gc * Nc;            // 47360 rows
constexpr int GNHc = GNc * Hc;          // 3,031,040 elems
constexpr int NHc = Nc * Hc;            // 23680

typedef __attribute__((ext_vector_type(16))) _Float16 v16h;
typedef __attribute__((ext_vector_type(8)))  float    v8f;

// ---------------- helpers ----------------
__device__ __forceinline__ float gelu_f(float x) {
    return 0.5f * x * (1.0f + erff(x * 0.70710678118654752f));
}
__device__ __forceinline__ float lrelu_f(float x) { return x > 0.f ? x : 0.2f * x; }

__device__ __forceinline__ float atomicMaxFloat(float* addr, float v) {
    // sign-split trick: works for any IEEE float given -inf init
    return (v >= 0.f)
        ? __int_as_float(atomicMax((int*)addr, __float_as_int(v)))
        : __uint_as_float(atomicMin((unsigned int*)addr, __float_as_uint(v)));
}
__device__ __forceinline__ float wave_sum(float v) {
    #pragma unroll
    for (int o = 16; o > 0; o >>= 1) v += __shfl_xor(v, o, 32);
    return v;
}

__device__ __forceinline__ v16h make_a_frag(const float* arow, int k0) {
    // lane<16 -> K {k0+0..15}; lane>=16 -> +16 (hs folded into arow)
    const float4* ap4 = reinterpret_cast<const float4*>(arow + k0);
    float4 a0 = ap4[0], a1 = ap4[1], a2 = ap4[2], a3 = ap4[3];
    v16h av;
    av[0]=(_Float16)a0.x; av[1]=(_Float16)a0.y; av[2]=(_Float16)a0.z; av[3]=(_Float16)a0.w;
    av[4]=(_Float16)a1.x; av[5]=(_Float16)a1.y; av[6]=(_Float16)a1.z; av[7]=(_Float16)a1.w;
    av[8]=(_Float16)a2.x; av[9]=(_Float16)a2.y; av[10]=(_Float16)a2.z; av[11]=(_Float16)a2.w;
    av[12]=(_Float16)a3.x; av[13]=(_Float16)a3.y; av[14]=(_Float16)a3.z; av[15]=(_Float16)a3.w;
    return av;
}

// ---------------- small utility kernels ----------------
__global__ void k_fill(float* p, float v, int n) {
    int i = blockIdx.x * blockDim.x + threadIdx.x;
    if (i < n) p[i] = v;
}
__global__ void k_copy4(float4* dst, const float4* src, int n4) {
    int i = blockIdx.x * blockDim.x + threadIdx.x;
    if (i < n4) dst[i] = src[i];
}
__global__ void k_bias_act(float* p, const float* __restrict__ b, int n, int nch, int act) {
    int i = blockIdx.x * blockDim.x + threadIdx.x;
    if (i >= n) return;
    float v = p[i] + b[i % nch];
    p[i] = act ? gelu_f(v) : v;
}

// ---------------- edge preprocessing ----------------
__global__ void k_edge_build(const int* ei, const float* ew,
                             int* src2, int* dst2, float* wbuf, float* deg) {
    int i = blockIdx.x * blockDim.x + threadIdx.x;
    if (i >= E2c) return;
    int s, d; float w;
    if (i < Ec) { s = ei[i]; d = ei[Ec + i]; w = ew[i]; }
    else        { s = d = i - Ec; w = 1.0f; }
    src2[i] = s; dst2[i] = d; wbuf[i] = w;
    atomicAdd(&deg[d], w);
}
__global__ void k_dis(float* degdis) {
    int i = blockIdx.x * blockDim.x + threadIdx.x;
    if (i >= Nc) return;
    float v = degdis[i];
    degdis[i] = v > 0.f ? rsqrtf(v) : 0.f;
}
__global__ void k_gnorm(const int* src2, const int* dst2, const float* wbuf,
                        const float* dis, float* gnorm) {
    int i = blockIdx.x * blockDim.x + threadIdx.x;
    if (i >= E2c) return;
    gnorm[i] = dis[src2[i]] * wbuf[i] * dis[dst2[i]];
}

// ---------------- WMMA GEMM, K=N=64 specialized: C(M,64) = A(M,64) @ B(64,64) ----
// One wave owns a 16x64 strip (4 f32 accumulators). B is staged once per block in
// LDS *pre-swizzled into the exact per-lane WMMA fragment layout*, so each operand
// is one contiguous 32B LDS read instead of 16 scalar ds_load_u16. K-loop fully
// unrolled: 8 back-to-back v_wmma, same-acc dependencies 4 ops apart.
__global__ void k_gemm64_wmma(const float* __restrict__ A,
                              const float* __restrict__ B,
                              float* __restrict__ C, int tiles_total) {
    __shared__ _Float16 bsh[4096];   // [kstep][nt][lane][j]  (8KB)
    for (int idx = threadIdx.x; idx < 4096; idx += blockDim.x) {
        int j     = idx & 15;
        int ln    = (idx >> 4) & 31;
        int slot  = idx >> 9;               // 0..7
        int nt    = slot & 3, kstep = slot >> 2;
        int kk    = kstep * 32 + ((ln >> 4) << 4);
        int col   = nt * 16 + (ln & 15);
        bsh[idx]  = (_Float16)B[(kk + j) * 64 + col];
    }
    __syncthreads();
    int wid = (blockIdx.x * blockDim.x + threadIdx.x) >> 5;
    if (wid >= tiles_total) return;
    int lane = threadIdx.x & 31;
    int m_base = wid * 16;
    int r  = lane & 15;
    int hs = lane >> 4;

    const v16h* bfr = reinterpret_cast<const v16h*>(bsh);
    const float* arow = A + (size_t)(m_base + r) * 64 + hs * 16;
    v8f cacc[4] = {{}, {}, {}, {}};
    #pragma unroll
    for (int kstep = 0; kstep < 2; ++kstep) {
        v16h av = make_a_frag(arow, kstep * 32);
        #pragma unroll
        for (int nt = 0; nt < 4; ++nt) {
            v16h bv = bfr[(kstep * 4 + nt) * 32 + lane];
            cacc[nt] = __builtin_amdgcn_wmma_f32_16x16x32_f16(false, av, false, bv,
                                                              (short)0, cacc[nt], false, false);
        }
    }
    // C layout: vgpr rr, lanes 0-15 -> M=rr, lanes 16-31 -> M=rr+8; N = lane&15
    #pragma unroll
    for (int nt = 0; nt < 4; ++nt) {
        int ccol = nt * 16 + (lane & 15);
        #pragma unroll
        for (int rr = 0; rr < 8; ++rr)
            C[(size_t)(m_base + rr + hs * 8) * 64 + ccol] = cacc[nt][rr];
    }
}

// ---------------- split-K WMMA GEMM (global B): partials via f32 atomics --------
// wave = one 16(M) x 64(N) strip for one K-slice; C pre-zeroed, bias/act later.
__global__ void k_gemm_wmma_sk(const float* __restrict__ A, int lda,
                               const float* __restrict__ B, int ldb,
                               float* __restrict__ C, int ldc,
                               int K, int ngrp, int tiles_total,
                               int ksplits, int kchunk) {
    int wid = (blockIdx.x * blockDim.x + threadIdx.x) >> 5;
    if (wid >= tiles_total) return;
    int lane = threadIdx.x & 31;
    int ks   = wid % ksplits;
    int tile = wid / ksplits;
    int mt = tile / ngrp, ng = tile % ngrp;
    int m_base = mt * 16, n_base = ng * 64;
    int r  = lane & 15;
    int hs = lane >> 4;

    v8f cacc[4] = {{}, {}, {}, {}};
    const float* arow = A + (size_t)(m_base + r) * lda + hs * 16;
    int kbeg = ks * kchunk, kend = kbeg + kchunk;
    for (int k0 = kbeg; k0 < kend && k0 < K; k0 += 32) {
        v16h av = make_a_frag(arow, k0);
        if (k0 + 32 < kend && k0 + 32 < K)
            __builtin_prefetch((const void*)(arow + k0 + 32), 0, 1);
        int kk = k0 + hs * 16;
        #pragma unroll
        for (int nt = 0; nt < 4; ++nt) {
            const float* bp = B + (size_t)kk * ldb + n_base + nt * 16 + r;
            v16h bv;
            #pragma unroll
            for (int j = 0; j < 16; ++j) bv[j] = (_Float16)bp[(size_t)j * ldb];
            cacc[nt] = __builtin_amdgcn_wmma_f32_16x16x32_f16(false, av, false, bv,
                                                              (short)0, cacc[nt], false, false);
        }
    }
    #pragma unroll
    for (int nt = 0; nt < 4; ++nt) {
        int ccol = n_base + nt * 16 + (lane & 15);
        #pragma unroll
        for (int rr = 0; rr < 8; ++rr)
            atomicAdd(&C[(size_t)(m_base + rr + hs * 8) * ldc + ccol], cacc[nt][rr]);
    }
}

// ---------------- GCN scatter ----------------
__global__ void k_gcn_scatter(const float* __restrict__ hw, float* __restrict__ g,
                              const int* src2, const int* dst2, const float* gnorm) {
    int e = blockIdx.x, gi = blockIdx.y, cch = threadIdx.x;
    float gn = gnorm[e];
    int s = src2[e], d = dst2[e];
    atomicAdd(&g[((size_t)gi * Nc + d) * Hc + cch],
              hw[((size_t)gi * Nc + s) * Hc + cch] * gn);
}

// ---------------- GAT attention ----------------
__global__ void k_att_scores(const float* __restrict__ gh,
                             const float* __restrict__ aw_s, const float* __restrict__ aw_d,
                             float* asrc, float* adst, int total) {
    int i = blockIdx.x * blockDim.x + threadIdx.x;
    if (i >= total) return;
    int row = i >> 2, head = i & 3;
    const float* gp = gh + (size_t)row * Hc + head * DHc;
    float s = 0.f, t = 0.f;
    #pragma unroll
    for (int d = 0; d < DHc; ++d) {
        float v = gp[d];
        s += v * aw_s[head * DHc + d];
        t += v * aw_d[head * DHc + d];
    }
    asrc[i] = s; adst[i] = t;
}
__global__ void k_att_max(const float* asrc, const float* adst,
                          const int* src2, const int* dst2, float* mmax) {
    int gi = blockIdx.y;
    int e = blockIdx.x * 32 + (threadIdx.x >> 2);
    int head = threadIdx.x & 3;
    if (e >= E2c) return;
    int s = src2[e], d = dst2[e];
    float ev = lrelu_f(asrc[((size_t)gi * Nc + s) * 4 + head] +
                       adst[((size_t)gi * Nc + d) * 4 + head]);
    atomicMaxFloat(&mmax[((size_t)gi * Nc + d) * 4 + head], ev);
}
__global__ void k_att_den(const float* asrc, const float* adst,
                          const int* src2, const int* dst2,
                          const float* mmax, float* den) {
    int gi = blockIdx.y;
    int e = blockIdx.x * 32 + (threadIdx.x >> 2);
    int head = threadIdx.x & 3;
    if (e >= E2c) return;
    int s = src2[e], d = dst2[e];
    size_t di = ((size_t)gi * Nc + d) * 4 + head;
    float ev = lrelu_f(asrc[((size_t)gi * Nc + s) * 4 + head] + adst[di]);
    atomicAdd(&den[di], expf(ev - mmax[di]));
}
__global__ void k_att_scatter(const float* __restrict__ gh,
                              const float* asrc, const float* adst,
                              const int* src2, const int* dst2,
                              const float* mmax, const float* den,
                              float* __restrict__ out) {
    int gi = blockIdx.y;
    int e = blockIdx.x * 32 + (threadIdx.x >> 2);
    int head = threadIdx.x & 3;
    if (e >= E2c) return;
    int s = src2[e], d = dst2[e];
    size_t di = ((size_t)gi * Nc + d) * 4 + head;
    float ev = lrelu_f(asrc[((size_t)gi * Nc + s) * 4 + head] + adst[di]);
    float alpha = expf(ev - mmax[di]) / den[di];
    const float* gp = gh + ((size_t)gi * Nc + s) * Hc + head * DHc;
    float* op = out + ((size_t)gi * Nc + d) * Hc + head * DHc;
    #pragma unroll
    for (int dd = 0; dd < DHc; ++dd) atomicAdd(&op[dd], gp[dd] * alpha);
}

// ---------------- residual + layernorm (one wave per 64-ch row) ----------------
__global__ void k_resid_ln(const float* __restrict__ att, const float* __restrict__ bg,
                           const float* __restrict__ lng, const float* __restrict__ lnb,
                           float* __restrict__ h) {
    int w = threadIdx.x >> 5, lane = threadIdx.x & 31;
    int row = blockIdx.x * 8 + w;
    int c0 = lane * 2;
    size_t base = (size_t)row * Hc;
    float r0 = att[base + c0]     + bg[c0]     + h[base + c0];
    float r1 = att[base + c0 + 1] + bg[c0 + 1] + h[base + c0 + 1];
    float mu = wave_sum(r0 + r1) * (1.0f / 64.0f);
    float d0 = r0 - mu, d1 = r1 - mu;
    float var = wave_sum(d0 * d0 + d1 * d1) * (1.0f / 64.0f);
    float rs = rsqrtf(var + 1e-5f);
    h[base + c0]     = d0 * rs * lng[c0]     + lnb[c0];
    h[base + c0 + 1] = d1 * rs * lng[c0 + 1] + lnb[c0 + 1];
}

// ---------------- layout transforms ----------------
__global__ void k_to_xt(const float* __restrict__ h, float* __restrict__ xt) {
    int i = blockIdx.x * blockDim.x + threadIdx.x;
    if (i >= GNHc) return;
    int bn = i >> 11, rem = i & 2047;
    int hc = rem >> 5, t = rem & 31;
    int b = bn / Nc, n = bn % Nc;
    xt[i] = h[(((size_t)(b * Tc + t) * Nc + n) << 6) + hc];
}
__global__ void k_to_z(const float* __restrict__ xt, float* __restrict__ z) {
    int i = blockIdx.x * blockDim.x + threadIdx.x;
    if (i >= GNHc) return;
    int row = i / NHc, col = i % NHc;
    int n = col >> 6, hc = col & 63;
    int b = row >> 5, t = row & 31;
    z[i] = xt[(((size_t)(b * Nc + n)) << 11) + (hc << 5) + t];
}

// ---------------- temporal conv: per (b,n), LDS tile of (64 ch x 32 t) ----------------
__global__ void k_tconv(const float* __restrict__ xin, float* __restrict__ xout,
                        const float* __restrict__ W, const float* __restrict__ bias) {
    __shared__ float tile[64][36];   // padded by (K-1)/2 on both sides (SAME, K=5)
    int bn = blockIdx.x, o = threadIdx.x;
    const float* xr = xin + (size_t)bn * 2048 + o * 32;
    tile[o][0] = 0.f; tile[o][1] = 0.f; tile[o][34] = 0.f; tile[o][35] = 0.f;
    #pragma unroll
    for (int t = 0; t < 32; ++t) tile[o][t + 2] = xr[t];
    __syncthreads();
    float acc[32];
    float bo = bias[o];
    #pragma unroll
    for (int t = 0; t < 32; ++t) acc[t] = bo;
    const float* wp = W + o * (Hc * Kc);
    for (int i = 0; i < 64; ++i) {
        float w0 = wp[i * 5 + 0], w1 = wp[i * 5 + 1], w2 = wp[i * 5 + 2];
        float w3 = wp[i * 5 + 3], w4 = wp[i * 5 + 4];
        #pragma unroll
        for (int t = 0; t < 32; ++t)
            acc[t] += w0 * tile[i][t] + w1 * tile[i][t + 1] + w2 * tile[i][t + 2]
                    + w3 * tile[i][t + 3] + w4 * tile[i][t + 4];
    }
    float* op = xout + (size_t)bn * 2048 + o * 32;
    #pragma unroll
    for (int t = 0; t < 32; ++t) op[t] = gelu_f(acc[t]) + tile[o][t + 2];
}

// ---------------- host orchestration ----------------
static inline unsigned cdiv(unsigned a, unsigned b) { return (a + b - 1) / b; }

extern "C" void kernel_launch(void* const* d_in, const int* in_sizes, int n_in,
                              void* d_out, int out_size, void* d_ws, size_t ws_size,
                              hipStream_t stream) {
    const float* x       = (const float*)d_in[0];
    const int*   eidx    = (const int*)  d_in[1];
    const float* ew      = (const float*)d_in[2];
    const float* W_gcn   = (const float*)d_in[3];
    const float* b_gcn   = (const float*)d_in[4];
    const float* W_gat   = (const float*)d_in[5];
    const float* att_src = (const float*)d_in[6];
    const float* att_dst = (const float*)d_in[7];
    const float* b_gat   = (const float*)d_in[8];
    const float* W_t     = (const float*)d_in[9];
    const float* b_t     = (const float*)d_in[10];
    const float* ln_g    = (const float*)d_in[11];
    const float* ln_b    = (const float*)d_in[12];
    const float* W1      = (const float*)d_in[13];
    const float* b1      = (const float*)d_in[14];
    const float* W2      = (const float*)d_in[15];
    const float* b2      = (const float*)d_in[16];
    float* out = (float*)d_out;

    // workspace carve-up (floats)
    float* ws   = (float*)d_ws;
    float* hcur = ws;
    float* tmpA = ws + (size_t)1 * GNHc;
    float* tmpB = ws + (size_t)2 * GNHc;
    float* xtA  = ws + (size_t)3 * GNHc;
    float* xtB  = ws + (size_t)4 * GNHc;
    float* sm   = ws + (size_t)5 * GNHc;
    float* gnorm  = sm; sm += 12288;
    float* wbuf   = sm; sm += 12288;
    float* degdis = sm; sm += 512;
    float* asrc   = sm; sm += (size_t)GNc * 4;
    float* adst   = sm; sm += (size_t)GNc * 4;
    float* mmax   = sm; sm += (size_t)GNc * 4;
    float* den    = sm; sm += (size_t)GNc * 4;
    float* h1     = sm; sm += 128 * 256;
    int* src2 = (int*)sm; sm += 12288;
    int* dst2 = (int*)sm; sm += 12288;

    // --- edge preprocessing ---
    k_fill<<<cdiv(Nc, 256), 256, 0, stream>>>(degdis, 0.f, Nc);
    k_edge_build<<<cdiv(E2c, 256), 256, 0, stream>>>(eidx, ew, src2, dst2, wbuf, degdis);
    k_dis<<<cdiv(Nc, 256), 256, 0, stream>>>(degdis);
    k_gnorm<<<cdiv(E2c, 256), 256, 0, stream>>>(src2, dst2, wbuf, degdis, gnorm);

    // h = x  (same flat layout (G,N,H))
    k_copy4<<<GNHc / 4 / 256, 256, 0, stream>>>((float4*)hcur, (const float4*)x, GNHc / 4);

    const int tilesGN = GNc / 16;                 // 2960 waves (16x64 strips)
    for (int l = 0; l < Lc; ++l) {
        const float* Wg = W_gcn + (size_t)l * Hc * Hc;
        const float* Wa = W_gat + (size_t)l * Hc * Hc;
        // hw = h @ W_gcn[l]      (swizzled-LDS 64x64 weights)
        k_gemm64_wmma<<<tilesGN / 4, 128, 0, stream>>>(hcur, Wg, tmpA, tilesGN);
        // g = scatter_add(msg) + b ; gelu
        k_fill<<<cdiv(GNHc, 256), 256, 0, stream>>>(tmpB, 0.f, GNHc);
        k_gcn_scatter<<<dim3(E2c, Gc), 64, 0, stream>>>(tmpA, tmpB, src2, dst2, gnorm);
        k_bias_act<<<cdiv(GNHc, 256), 256, 0, stream>>>(tmpB, b_gcn + l * Hc, GNHc, Hc, 1);
        // gh = g @ W_gat[l]
        k_gemm64_wmma<<<tilesGN / 4, 128, 0, stream>>>(tmpB, Wa, tmpA, tilesGN);
        // attention scores
        k_att_scores<<<cdiv(GNc * 4, 256), 256, 0, stream>>>(
            tmpA, att_src + l * HEADSc * DHc, att_dst + l * HEADSc * DHc,
            asrc, adst, GNc * 4);
        k_fill<<<cdiv(GNc * 4, 256), 256, 0, stream>>>(mmax, -INFINITY, GNc * 4);
        k_fill<<<cdiv(GNc * 4, 256), 256, 0, stream>>>(den, 0.f, GNc * 4);
        dim3 egrid(cdiv(E2c, 32), Gc);
        k_att_max<<<egrid, 128, 0, stream>>>(asrc, adst, src2, dst2, mmax);
        k_att_den<<<egrid, 128, 0, stream>>>(asrc, adst, src2, dst2, mmax, den);
        k_fill<<<cdiv(GNHc, 256), 256, 0, stream>>>(tmpB, 0.f, GNHc);
        k_att_scatter<<<egrid, 128, 0, stream>>>(tmpA, asrc, adst, src2, dst2,
                                                 mmax, den, tmpB);
        // residual + layernorm -> hcur
        k_resid_ln<<<GNc / 8, 256, 0, stream>>>(tmpB, b_gat + l * Hc,
                                                ln_g + l * Hc, ln_b + l * Hc, hcur);
    }

    // temporal convs over T per (b,n)
    k_to_xt<<<cdiv(GNHc, 256), 256, 0, stream>>>(hcur, xtA);
    for (int l = 0; l < Lc; ++l) {
        const float* xin  = (l & 1) ? xtB : xtA;
        float*       xout = (l & 1) ? xtA : xtB;
        k_tconv<<<Bc * Nc, 64, 0, stream>>>(xin, xout,
                                            W_t + (size_t)l * Hc * Hc * Kc,
                                            b_t + l * Hc);
    }
    // final MLP: z(128,23680) -> gelu(z@W1+b1)(128,256) -> @W2+b2 (128,64)
    k_to_z<<<cdiv(GNHc, 256), 256, 0, stream>>>(xtB, tmpA);
    // split-K GEMM 1: 8 m-tiles x 4 col-groups x 37 K-slices (740 = 37*20 k-steps)
    k_fill<<<cdiv(128 * 256, 256), 256, 0, stream>>>(h1, 0.f, 128 * 256);
    k_gemm_wmma_sk<<<(8 * 4 * 37) / 4, 128, 0, stream>>>(
        tmpA, NHc, W1, 256, h1, 256, NHc, 4, 8 * 4 * 37, 37, 640);
    k_bias_act<<<cdiv(128 * 256, 256), 256, 0, stream>>>(h1, b1, 128 * 256, 256, 1);
    // split-K GEMM 2: 8 m-tiles x 1 col-group x 8 K-slices (256 = 8*32)
    k_fill<<<cdiv(Gc * Hc, 256), 256, 0, stream>>>(out, 0.f, Gc * Hc);
    k_gemm_wmma_sk<<<(8 * 1 * 8) / 4, 128, 0, stream>>>(
        h1, 256, W2, 64, out, 64, 256, 1, 8 * 1 * 8, 8, 32);
    k_bias_act<<<cdiv(Gc * Hc, 256), 256, 0, stream>>>(out, b2, Gc * Hc, 64, 0);
}
